// GAT2_33217277067704
// MI455X (gfx1250) — matrix operand order repaired and checked by reading, not compiled, not run on hardware
//
#include <hip/hip_runtime.h>

#define N_NODES   10000
#define E_EDGES   160000
#define HEADS     4
#define IN_CH     128
#define EDGE_DIM  16
#define NUM_GRAPHS 64
#define C1        512
#define C2        64
#define HC1       (HEADS * C1)   // 2048
#define HC2       (HEADS * C2)   // 256

typedef __attribute__((ext_vector_type(16))) __bf16 v16bf;
typedef __attribute__((ext_vector_type(8)))  __bf16 v8bf;
typedef __attribute__((ext_vector_type(8)))  float  v8f;

static inline int cdiv(int a, int b) { return (a + b - 1) / b; }

// ---------------- elementwise / setup kernels ----------------

__global__ void k_f32_to_bf16(const float* __restrict__ in, __bf16* __restrict__ out, int n) {
  int i = blockIdx.x * blockDim.x + threadIdx.x;
  if (i < n) out[i] = (__bf16)in[i];
}

// W[k][n] fp32 (KxN) -> Wt[n][k] bf16 (NxK)
__global__ void k_transpose_bf16(const float* __restrict__ W, __bf16* __restrict__ Wt, int K, int Nc) {
  int i = blockIdx.x * blockDim.x + threadIdx.x;
  if (i < K * Nc) {
    int k = i / Nc, n = i - k * Nc;
    Wt[(size_t)n * K + k] = (__bf16)W[i];
  }
}

__global__ void k_fill_f32(float* __restrict__ p, float v, int n) {
  int i = blockIdx.x * blockDim.x + threadIdx.x;
  if (i < n) p[i] = v;
}

__global__ void k_fill_u32(unsigned* __restrict__ p, unsigned v, int n) {
  int i = blockIdx.x * blockDim.x + threadIdx.x;
  if (i < n) p[i] = v;
}

// out[node*HC + c] = bias[c]
__global__ void k_init_bias(float* __restrict__ out, const float* __restrict__ bias, int HC, int total) {
  int i = blockIdx.x * blockDim.x + threadIdx.x;
  if (i < total) out[i] = bias[i % HC];
}

// relu in place, optional bf16 copy-out
__global__ void k_relu_bf16(float* __restrict__ x, __bf16* __restrict__ xb, int n) {
  int i = blockIdx.x * blockDim.x + threadIdx.x;
  if (i < n) {
    float v = x[i];
    v = v > 0.f ? v : 0.f;
    x[i] = v;
    if (xb) xb[i] = (__bf16)v;
  }
}

// ---------------- WMMA GEMM: C[M,N] = A[M,K](bf16) @ Bt[N,K](bf16) + bias ----------------
// One wave computes a 16x16 tile with v_wmma_f32_16x16x32_bf16.

__global__ void k_gemm_bf16_wmma(const __bf16* __restrict__ A,
                                 const __bf16* __restrict__ Bt,
                                 const float* __restrict__ bias,
                                 __bf16* __restrict__ Cb,   // bf16 out (nullable)
                                 float* __restrict__ Cf,    // fp32 out (nullable)
                                 int M, int Nc, int K) {
  const int lane   = threadIdx.x & 31;
  const int wave   = blockIdx.x * (blockDim.x >> 5) + (threadIdx.x >> 5);
  const int tilesN = Nc >> 4;
  const int tilesM = (M + 15) >> 4;
  const int tM = wave / tilesN;
  const int tN = wave - tM * tilesN;
  if (tM >= tilesM) return;  // wave-uniform exit: EXEC stays all-ones for WMMA

  const int rl    = lane & 15;
  const int khalf = lane >> 4;
  const int aRow  = min(tM * 16 + rl, M - 1);
  const __bf16* Ap = A  + (size_t)aRow * K + khalf * 8;
  const __bf16* Bp = Bt + (size_t)(tN * 16 + rl) * K + khalf * 8;

  v8f acc = {};
  for (int k0 = 0; k0 < K; k0 += 32) {
    // A fragment (16-bit A 16x32 layout): lane rl/khalf holds K = khalf*8 + j (j<8)
    // and K = 16 + khalf*8 + (j-8) (j>=8) -> two contiguous 16B loads.
    v8bf a0 = *(const v8bf*)(Ap + k0);
    v8bf a1 = *(const v8bf*)(Ap + k0 + 16);
    // B fragment: lane holds column tN*16+rl, same K order as A -> two contiguous 16B loads in Bt.
    v8bf b0 = *(const v8bf*)(Bp + k0);
    v8bf b1 = *(const v8bf*)(Bp + k0 + 16);
    v16bf a, b;
#pragma unroll
    for (int j = 0; j < 8; ++j) {
      a[j] = a0[j]; a[j + 8] = a1[j];
      b[j] = b0[j]; b[j + 8] = b1[j];
    }
    acc = __builtin_amdgcn_wmma_f32_16x16x32_bf16(false, a, false, b, (short)0, acc, false, false);
  }

  const int colg = tN * 16 + rl;
  const float bv = bias ? bias[colg] : 0.0f;
#pragma unroll
  for (int i = 0; i < 8; ++i) {
    int mrow = tM * 16 + khalf * 8 + i;  // C/D layout: VGPR i -> M = i + 8*khalf
    if (mrow < M) {
      float v = acc[i] + bv;
      if (Cb) Cb[(size_t)mrow * Nc + colg] = (__bf16)v;
      if (Cf) Cf[(size_t)mrow * Nc + colg] = v;
    }
  }
}

// ---------------- edge logits: a^T LeakyReLU(xl[src] + xr[dst] + edge_attr@We) ----------------
// One wave per edge (grid-strided); We staged in LDS as bf16 (max 16*2048*2 = 64KB).

__global__ void k_edge_logits(const __bf16* __restrict__ xl, const __bf16* __restrict__ xr,
                              const float* __restrict__ edge_attr,
                              const float* __restrict__ We,   // [16, HC] fp32
                              const float* __restrict__ att,  // [HC] (H*C flattened)
                              const int* __restrict__ src, const int* __restrict__ dst,
                              float* __restrict__ logits, int E, int HC, int C) {
  __shared__ __bf16 sWe[EDGE_DIM * HC1];
  for (int i = threadIdx.x; i < EDGE_DIM * HC; i += blockDim.x) sWe[i] = (__bf16)We[i];
  __syncthreads();

  const int lane   = threadIdx.x & 31;
  const int wave   = blockIdx.x * (blockDim.x >> 5) + (threadIdx.x >> 5);
  const int nwaves = gridDim.x * (blockDim.x >> 5);

  for (int e = wave; e < E; e += nwaves) {
    const int s = src[e], d = dst[e];
    float ea[EDGE_DIM];
#pragma unroll
    for (int k = 0; k < EDGE_DIM; ++k) ea[k] = edge_attr[e * EDGE_DIM + k];
    const __bf16* xls = xl + (size_t)s * HC;
    const __bf16* xrd = xr + (size_t)d * HC;

    for (int h = 0; h < HEADS; ++h) {
      float acc = 0.f;
      for (int cc = 0; cc < C; cc += 32) {
        int c = h * C + cc + lane;
        float v = (float)xls[c] + (float)xrd[c];
#pragma unroll
        for (int k = 0; k < EDGE_DIM; ++k) v += ea[k] * (float)sWe[k * HC + c];
        v = v > 0.f ? v : 0.2f * v;  // leaky_relu(0.2)
        acc += v * att[c];
      }
#pragma unroll
      for (int off = 16; off > 0; off >>= 1) acc += __shfl_xor(acc, off, 32);
      if (lane == 0) logits[e * HEADS + h] = acc;
    }
  }
}

// ---------------- segment softmax over dst ----------------

__device__ __forceinline__ unsigned fmap(float f) {
  unsigned u = __float_as_uint(f);
  return (u & 0x80000000u) ? ~u : (u | 0x80000000u);  // order-preserving float->uint
}
__device__ __forceinline__ float funmap(unsigned u) {
  return (u & 0x80000000u) ? __uint_as_float(u & 0x7fffffffu) : __uint_as_float(~u);
}

__global__ void k_segmax(const float* __restrict__ logits, const int* __restrict__ dst,
                         unsigned* __restrict__ segmax, int E) {
  int i = blockIdx.x * blockDim.x + threadIdx.x;
  if (i < E * HEADS) {
    int e = i >> 2, h = i & 3;
    atomicMax(&segmax[dst[e] * HEADS + h], fmap(logits[i]));
  }
}

__global__ void k_expsum(const float* __restrict__ logits, const int* __restrict__ dst,
                         const unsigned* __restrict__ segmax,
                         float* __restrict__ alpha, float* __restrict__ segsum, int E) {
  int i = blockIdx.x * blockDim.x + threadIdx.x;
  if (i < E * HEADS) {
    int e = i >> 2, h = i & 3;
    int di = dst[e] * HEADS + h;
    float m = funmap(segmax[di]);
    if ((__float_as_uint(m) & 0x7f800000u) == 0x7f800000u) m = 0.f;  // non-finite -> 0
    float ex = __expf(logits[i] - m);
    alpha[i] = ex;
    atomicAdd(&segsum[di], ex);
  }
}

// ---------------- scatter alpha * xl[src] into out[dst] ----------------

__global__ void k_scatter(const __bf16* __restrict__ xl,
                          const int* __restrict__ src, const int* __restrict__ dst,
                          const float* __restrict__ alpha, const float* __restrict__ segsum,
                          float* __restrict__ out, int E, int HC, int C) {
  const int lane   = threadIdx.x & 31;
  const int wave   = blockIdx.x * (blockDim.x >> 5) + (threadIdx.x >> 5);
  const int nwaves = gridDim.x * (blockDim.x >> 5);
  for (int e = wave; e < E; e += nwaves) {
    const int s = src[e], d = dst[e];
    const __bf16* xls = xl + (size_t)s * HC;
    float* od = out + (size_t)d * HC;
    for (int h = 0; h < HEADS; ++h) {
      float a = alpha[e * HEADS + h] / (segsum[d * HEADS + h] + 1e-16f);
      for (int cc = 0; cc < C; cc += 32) {
        int c = h * C + cc + lane;
        atomicAdd(&od[c], a * (float)xls[c]);
      }
    }
  }
}

// ---------------- pooling + MLP head ----------------

__global__ void k_pool(const float* __restrict__ h2, const int* __restrict__ batch,
                       float* __restrict__ pooled, int total) {
  int i = blockIdx.x * blockDim.x + threadIdx.x;
  if (i < total) {
    int node = i / HC2, c = i - node * HC2;
    atomicAdd(&pooled[batch[node] * HC2 + c], h2[i]);
  }
}

__global__ void k_fc1(const float* __restrict__ pooled, const float* __restrict__ w,
                      const float* __restrict__ b, float* __restrict__ o) {
  int g = blockIdx.x, j = threadIdx.x;  // 64 graphs x 64 outputs
  float acc = b[j];
  for (int k = 0; k < HC2; ++k) acc += pooled[g * HC2 + k] * w[k * 64 + j];
  o[g * 64 + j] = acc > 0.f ? acc : 0.f;
}

__global__ void k_fc2(const float* __restrict__ t, const float* __restrict__ w,
                      const float* __restrict__ b, float* __restrict__ out) {
  int g = threadIdx.x;  // 64 graphs
  float acc = b[0];
  for (int k = 0; k < 64; ++k) acc += t[g * 64 + k] * w[k];
  out[g] = acc;
}

// ---------------- host orchestration ----------------

extern "C" void kernel_launch(void* const* d_in, const int* in_sizes, int n_in,
                              void* d_out, int out_size, void* d_ws, size_t ws_size,
                              hipStream_t stream) {
  const float* x        = (const float*)d_in[0];
  const int*   eidx     = (const int*)d_in[1];
  const float* eattr    = (const float*)d_in[2];
  const int*   batch    = (const int*)d_in[3];
  const float* Wl1 = (const float*)d_in[4];  const float* bl1 = (const float*)d_in[5];
  const float* Wr1 = (const float*)d_in[6];  const float* br1 = (const float*)d_in[7];
  const float* We1 = (const float*)d_in[8];  const float* att1 = (const float*)d_in[9];
  const float* bias1 = (const float*)d_in[10];
  const float* Wl2 = (const float*)d_in[11]; const float* bl2 = (const float*)d_in[12];
  const float* Wr2 = (const float*)d_in[13]; const float* br2 = (const float*)d_in[14];
  const float* We2 = (const float*)d_in[15]; const float* att2 = (const float*)d_in[16];
  const float* bias2 = (const float*)d_in[17];
  const float* fc1_w = (const float*)d_in[18]; const float* fc1_b = (const float*)d_in[19];
  const float* fc2_w = (const float*)d_in[20]; const float* fc2_b = (const float*)d_in[21];

  const int* src = eidx;             // edge_index[0]
  const int* dst = eidx + E_EDGES;   // edge_index[1]

  // ---- workspace carve-out (256B aligned), buffers reused across layers ----
  char* ws = (char*)d_ws;
  size_t off = 0;
  auto carve = [&](size_t bytes) -> char* {
    char* p = ws + off;
    off += (bytes + 255) & ~(size_t)255;
    return p;
  };
  __bf16* A_bf   = (__bf16*)carve((size_t)N_NODES * HC1 * 2);  // x bf16 / h bf16
  __bf16* Wt_l   = (__bf16*)carve((size_t)2048 * 256 * 2);     // max(128*2048, 2048*256)
  __bf16* Wt_r   = (__bf16*)carve((size_t)2048 * 256 * 2);
  __bf16* xl_bf  = (__bf16*)carve((size_t)N_NODES * HC1 * 2);
  __bf16* xr_bf  = (__bf16*)carve((size_t)N_NODES * HC1 * 2);
  float*  hbuf   = (float*) carve((size_t)N_NODES * HC1 * 4);  // layer out (fp32)
  float*  logits = (float*) carve((size_t)E_EDGES * HEADS * 4);
  float*  alpha  = (float*) carve((size_t)E_EDGES * HEADS * 4);
  unsigned* smax = (unsigned*)carve((size_t)N_NODES * HEADS * 4);
  float*  ssum   = (float*) carve((size_t)N_NODES * HEADS * 4);
  float*  pooled = (float*) carve((size_t)NUM_GRAPHS * HC2 * 4);
  float*  fc1o   = (float*) carve((size_t)NUM_GRAPHS * 64 * 4);
  (void)ws_size; (void)in_sizes; (void)n_in; (void)out_size;

  const int T = 256;
  auto g1 = [&](int n) { return dim3(cdiv(n, T)); };

  // ===================== Layer 1 =====================
  k_f32_to_bf16<<<g1(N_NODES * IN_CH), T, 0, stream>>>(x, A_bf, N_NODES * IN_CH);
  k_transpose_bf16<<<g1(IN_CH * HC1), T, 0, stream>>>(Wl1, Wt_l, IN_CH, HC1);
  k_transpose_bf16<<<g1(IN_CH * HC1), T, 0, stream>>>(Wr1, Wt_r, IN_CH, HC1);

  {
    int tiles = cdiv(N_NODES, 16) * (HC1 / 16);        // 625 * 128
    int blocks = cdiv(tiles, T / 32);
    k_gemm_bf16_wmma<<<blocks, T, 0, stream>>>(A_bf, Wt_l, bl1, xl_bf, nullptr, N_NODES, HC1, IN_CH);
    k_gemm_bf16_wmma<<<blocks, T, 0, stream>>>(A_bf, Wt_r, br1, xr_bf, nullptr, N_NODES, HC1, IN_CH);
  }

  k_fill_u32<<<g1(N_NODES * HEADS), T, 0, stream>>>(smax, 0u, N_NODES * HEADS);
  k_fill_f32<<<g1(N_NODES * HEADS), T, 0, stream>>>(ssum, 0.f, N_NODES * HEADS);

  k_edge_logits<<<1024, T, 0, stream>>>(xl_bf, xr_bf, eattr, We1, att1, src, dst,
                                        logits, E_EDGES, HC1, C1);
  k_segmax<<<g1(E_EDGES * HEADS), T, 0, stream>>>(logits, dst, smax, E_EDGES);
  k_expsum<<<g1(E_EDGES * HEADS), T, 0, stream>>>(logits, dst, smax, alpha, ssum, E_EDGES);

  k_init_bias<<<g1(N_NODES * HC1), T, 0, stream>>>(hbuf, bias1, HC1, N_NODES * HC1);
  k_scatter<<<1024, T, 0, stream>>>(xl_bf, src, dst, alpha, ssum, hbuf, E_EDGES, HC1, C1);
  k_relu_bf16<<<g1(N_NODES * HC1), T, 0, stream>>>(hbuf, A_bf, N_NODES * HC1);  // h -> bf16 for layer 2

  // ===================== Layer 2 =====================
  k_transpose_bf16<<<g1(HC1 * HC2), T, 0, stream>>>(Wl2, Wt_l, HC1, HC2);
  k_transpose_bf16<<<g1(HC1 * HC2), T, 0, stream>>>(Wr2, Wt_r, HC1, HC2);

  {
    int tiles = cdiv(N_NODES, 16) * (HC2 / 16);        // 625 * 16
    int blocks = cdiv(tiles, T / 32);
    k_gemm_bf16_wmma<<<blocks, T, 0, stream>>>(A_bf, Wt_l, bl2, xl_bf, nullptr, N_NODES, HC2, HC1);
    k_gemm_bf16_wmma<<<blocks, T, 0, stream>>>(A_bf, Wt_r, br2, xr_bf, nullptr, N_NODES, HC2, HC1);
  }

  k_fill_u32<<<g1(N_NODES * HEADS), T, 0, stream>>>(smax, 0u, N_NODES * HEADS);
  k_fill_f32<<<g1(N_NODES * HEADS), T, 0, stream>>>(ssum, 0.f, N_NODES * HEADS);

  k_edge_logits<<<1024, T, 0, stream>>>(xl_bf, xr_bf, eattr, We2, att2, src, dst,
                                        logits, E_EDGES, HC2, C2);
  k_segmax<<<g1(E_EDGES * HEADS), T, 0, stream>>>(logits, dst, smax, E_EDGES);
  k_expsum<<<g1(E_EDGES * HEADS), T, 0, stream>>>(logits, dst, smax, alpha, ssum, E_EDGES);

  k_init_bias<<<g1(N_NODES * HC2), T, 0, stream>>>(hbuf, bias2, HC2, N_NODES * HC2);
  k_scatter<<<1024, T, 0, stream>>>(xl_bf, src, dst, alpha, ssum, hbuf, E_EDGES, HC2, C2);
  k_relu_bf16<<<g1(N_NODES * HC2), T, 0, stream>>>(hbuf, nullptr, N_NODES * HC2);

  // ===================== Pool + MLP =====================
  k_fill_f32<<<g1(NUM_GRAPHS * HC2), T, 0, stream>>>(pooled, 0.f, NUM_GRAPHS * HC2);
  k_pool<<<g1(N_NODES * HC2), T, 0, stream>>>(hbuf, batch, pooled, N_NODES * HC2);
  k_fc1<<<NUM_GRAPHS, 64, 0, stream>>>(pooled, fc1_w, fc1_b, fc1o);
  k_fc2<<<1, NUM_GRAPHS, 0, stream>>>(fc1o, fc2_w, fc2_b, (float*)d_out);
}
